// GSEDroidModel_317827580076
// MI455X (gfx1250) — matrix operand order repaired and checked by reading, not compiled, not run on hardware
//
#include <hip/hip_runtime.h>
#include <hip/hip_bf16.h>

typedef __attribute__((ext_vector_type(16))) _Float16 v16h;
typedef __attribute__((ext_vector_type(8)))  float    v8f;

#define HID 128
#define NPGC 2048

// ---------------- utility ----------------
__global__ void k_fill(float* __restrict__ p, float v, long n) {
  long i = (long)blockIdx.x * blockDim.x + threadIdx.x;
  if (i < n) p[i] = v;
}

// Pack a 128x128 f32 weight matrix into f16 "B-fragment" order:
// P[((tile*4 + ks)*32 + lane)*16 + t] = W[(ks*32 + (lane>>4)*16 + t)*128 + tile*16 + (lane&15)]
// so each wave reads one whole fragment as a contiguous 32-byte v16h.
__global__ void k_pack_w(const float* __restrict__ W, _Float16* __restrict__ P) {
  int id = blockIdx.x * blockDim.x + threadIdx.x;   // 16384 threads
  int t    = id & 15;
  int lane = (id >> 4) & 31;
  int ks   = (id >> 9) & 3;
  int tile = id >> 11;
  int n = tile * 16 + (lane & 15);
  int k = ks * 32 + (lane >> 4) * 16 + t;
  P[id] = (_Float16)W[k * HID + n];
}

// ---------------- edge aggregation: agg[dst] += h[src]*em; cnt[dst] += em ----
__global__ void k_aggregate(const float* __restrict__ h,
                            const int* __restrict__ src,
                            const int* __restrict__ dst,
                            const float* __restrict__ em,
                            float* __restrict__ agg,
                            float* __restrict__ cnt,
                            int E, int withCnt) {
  long gid = (long)blockIdx.x * blockDim.x + threadIdx.x;
  int e = (int)(gid >> 5);
  if (e >= E) return;
  float w = em[e];
  if (w == 0.f) return;
  int part = (int)(gid & 31);
  int s = src[e], d = dst[e];
  const float4 v = *(const float4*)(h + (size_t)s * HID + part * 4);
  float* o = agg + (size_t)d * HID + part * 4;
  atomicAdd(o + 0, v.x * w);
  atomicAdd(o + 1, v.y * w);
  atomicAdd(o + 2, v.z * w);
  atomicAdd(o + 3, v.w * w);
  if (withCnt && part == 0) atomicAdd(cnt + d, w);
}

__global__ void k_mean(float* __restrict__ agg, const float* __restrict__ cnt, long n) {
  long gid = (long)blockIdx.x * blockDim.x + threadIdx.x;
  if (gid >= n) return;
  float c = cnt[gid >> 7];
  c = (c < 1.f) ? 1.f : c;
  agg[gid] = agg[gid] / c;
}

// ---------------- fused SAGE GEMM (WMMA f16 -> f32) -------------------------
// out = relu( Am @ Wl + Ah @ Wr + bias ) * nmask   , all rows Nt x 128
// grid = Nt/16 blocks of 256 threads (8 waves). Wave w -> N-tile [16w,16w+16).
// Weights pre-packed to f16 fragment order (see k_pack_w).
__global__ void __launch_bounds__(256)
k_sage_gemm(const float* __restrict__ Am, const float* __restrict__ Ah,
            const _Float16* __restrict__ WlPk, const _Float16* __restrict__ WrPk,
            const float* __restrict__ bias, const float* __restrict__ nmask,
            float* __restrict__ out) {
  const int lane = threadIdx.x & 31;
  const int wid  = threadIdx.x >> 5;
  const int n0   = wid * 16;
  const int m0   = blockIdx.x * 16;
  const int lm   = lane & 15;
  const int lh   = lane >> 4;
  const int row  = m0 + lm;

  const float* Arow = Am + (size_t)row * HID;
  const float* Hrow = Ah + (size_t)row * HID;
  const size_t fragBase = (((size_t)wid * 4) * 32 + lane) * 16;   // + ks*32*16

  v8f acc = {};
  // pass 1: mean-aggregated neighbors @ Wl
  #pragma unroll
  for (int ks = 0; ks < 4; ++ks) {
    const int kb = ks * 32;
    v16h a;
    #pragma unroll
    for (int t = 0; t < 16; ++t) {
      int k = kb + lh * 16 + t;   // A frag (16x32 f16)
      a[t] = (_Float16)Arow[k];
    }
    const v16h b = *(const v16h*)(WlPk + fragBase + (size_t)ks * 512);
    acc = __builtin_amdgcn_wmma_f32_16x16x32_f16(false, a, false, b,
                                                 (short)0, acc, false, false);
  }
  // pass 2: root features @ Wr
  #pragma unroll
  for (int ks = 0; ks < 4; ++ks) {
    const int kb = ks * 32;
    v16h a;
    #pragma unroll
    for (int t = 0; t < 16; ++t) {
      int k = kb + lh * 16 + t;
      a[t] = (_Float16)Hrow[k];
    }
    const v16h b = *(const v16h*)(WrPk + fragBase + (size_t)ks * 512);
    acc = __builtin_amdgcn_wmma_f32_16x16x32_f16(false, a, false, b,
                                                 (short)0, acc, false, false);
  }

  const int n  = n0 + lm;
  const float bn = bias[n];
  #pragma unroll
  for (int r = 0; r < 8; ++r) {
    int m = m0 + r + lh * 8;            // C frag: lanes 0-15 -> M=r, 16-31 -> M=8+r
    float v = acc[r] + bn;
    v = v > 0.f ? v : 0.f;
    v *= nmask[m];
    out[(size_t)m * HID + n] = v;
  }
}

// ---------------- SAGPool score: score[i] = agg[i].Prel + h[i].Proot + prb ---
__global__ void k_score(const float* __restrict__ agg, const float* __restrict__ h,
                        const float* __restrict__ Prel, const float* __restrict__ Proot,
                        const float* __restrict__ prb, float* __restrict__ score, int Nt) {
  long gid = (long)blockIdx.x * blockDim.x + threadIdx.x;
  int node = (int)(gid >> 5);
  int lane = threadIdx.x & 31;
  if (node >= Nt) return;
  const float* ar = agg + (size_t)node * HID;
  const float* hr = h + (size_t)node * HID;
  float p = 0.f;
  #pragma unroll
  for (int t = 0; t < 4; ++t) {
    int c = lane * 4 + t;
    p += ar[c] * Prel[c] + hr[c] * Proot[c];
  }
  #pragma unroll
  for (int off = 16; off > 0; off >>= 1) p += __shfl_down(p, off, 32);
  if (lane == 0) score[node] = p + prb[0];
}

// ---------------- top-k per graph via in-LDS bitonic sort -------------------
__device__ inline bool gt_pair(float a, int ia, float b, int ib) {
  return (a > b) || (a == b && ia < ib);   // descending score, ascending index
}

__global__ void __launch_bounds__(1024)
k_topk(const float* __restrict__ score, float* __restrict__ nm, int k) {
  __shared__ float sk[NPGC];
  __shared__ int   si[NPGC];
  const int g = blockIdx.x, tid = threadIdx.x;
  const int base = g * NPGC;
  for (int j = tid; j < NPGC; j += 1024) {
    float m = nm[base + j];
    sk[j] = (m > 0.f) ? score[base + j] : -1e30f;
    si[j] = j;
  }
  __syncthreads();
  for (int j = tid; j < NPGC; j += 1024) nm[base + j] = 0.f;  // block owns range

  // bitonic build
  for (int size = 2; size < NPGC; size <<= 1) {
    const bool ddd = ((tid & (size >> 1)) != 0);
    for (int stride = size >> 1; stride > 0; stride >>= 1) {
      __syncthreads();
      int pos = 2 * tid - (tid & (stride - 1));
      float ka = sk[pos], kb = sk[pos + stride];
      int   ia = si[pos], ib = si[pos + stride];
      if (gt_pair(ka, ia, kb, ib) == ddd) {
        sk[pos] = kb; sk[pos + stride] = ka;
        si[pos] = ib; si[pos + stride] = ia;
      }
    }
  }
  // final merge: descending
  for (int stride = NPGC >> 1; stride > 0; stride >>= 1) {
    __syncthreads();
    int pos = 2 * tid - (tid & (stride - 1));
    float ka = sk[pos], kb = sk[pos + stride];
    int   ia = si[pos], ib = si[pos + stride];
    if (!gt_pair(ka, ia, kb, ib)) {
      sk[pos] = kb; sk[pos + stride] = ka;
      si[pos] = ib; si[pos + stride] = ia;
    }
  }
  __syncthreads();
  for (int j = tid; j < NPGC; j += 1024)
    if (j < k) nm[base + si[j]] = 1.f;
}

// ---------------- pool application ------------------------------------------
__global__ void k_pool_h(float* __restrict__ h, const float* __restrict__ score,
                         const float* __restrict__ nm, long n) {
  long gid = (long)blockIdx.x * blockDim.x + threadIdx.x;
  if (gid >= n) return;
  long i = gid >> 7;
  h[gid] = h[gid] * tanhf(score[i]) * nm[i];
}

__global__ void k_pool_em(float* __restrict__ em, const int* __restrict__ src,
                          const int* __restrict__ dst, const float* __restrict__ nm, int E) {
  int e = blockIdx.x * blockDim.x + threadIdx.x;
  if (e >= E) return;
  em[e] = em[e] * nm[src[e]] * nm[dst[e]];
}

// ---------------- global mean pool ------------------------------------------
__global__ void k_readout(const float* __restrict__ h, const float* __restrict__ nm,
                          float* __restrict__ g) {
  __shared__ float cs[HID];
  const int gr = blockIdx.x, c = threadIdx.x;   // 128 threads
  const size_t base = (size_t)gr * NPGC;
  float s = 0.f;
  for (int r = 0; r < NPGC; ++r) s += h[(base + r) * HID + c];
  float cn = 0.f;
  for (int r = c; r < NPGC; r += HID) cn += nm[base + r];
  cs[c] = cn;
  __syncthreads();
  for (int off = 64; off > 0; off >>= 1) {
    if (c < off) cs[c] += cs[c + off];
    __syncthreads();
  }
  float cnt = cs[0];
  cnt = cnt < 1.f ? 1.f : cnt;
  g[gr * HID + c] = s / cnt;
}

// ---------------- MLP head + log_softmax ------------------------------------
__global__ void k_head(const float* __restrict__ g, const float* __restrict__ W5,
                       const float* __restrict__ b5, const float* __restrict__ W6,
                       const float* __restrict__ b6, float* __restrict__ out) {
  __shared__ float g5[64];
  const int gr = blockIdx.x, j = threadIdx.x;   // 64 threads
  float acc = b5[j];
  for (int c = 0; c < HID; ++c) acc += g[gr * HID + c] * W5[c * 64 + j];
  g5[j] = acc > 0.f ? acc : 0.f;
  __syncthreads();
  if (j == 0) {
    float l0 = b6[0], l1 = b6[1];
    for (int c = 0; c < 64; ++c) { l0 += g5[c] * W6[c * 2 + 0]; l1 += g5[c] * W6[c * 2 + 1]; }
    float m = l0 > l1 ? l0 : l1;
    float lse = m + logf(expf(l0 - m) + expf(l1 - m));
    out[gr * 2 + 0] = l0 - lse;
    out[gr * 2 + 1] = l1 - lse;
  }
}

// ---------------- driver -----------------------------------------------------
static inline int cdiv(long a, long b) { return (int)((a + b - 1) / b); }

extern "C" void kernel_launch(void* const* d_in, const int* in_sizes, int n_in,
                              void* d_out, int out_size, void* d_ws, size_t ws_size,
                              hipStream_t stream) {
  (void)n_in; (void)out_size; (void)ws_size;
  const int Nt = in_sizes[0] / HID;
  const int E  = in_sizes[1] / 2;
  const int G  = Nt / NPGC;
  const long NtF = (long)Nt * HID;

  const float* x   = (const float*)d_in[0];
  const int*   src = (const int*)d_in[1];
  const int*   dst = src + E;
  const float* Wl1 = (const float*)d_in[4];
  const float* bl1 = (const float*)d_in[5];
  const float* Wr1 = (const float*)d_in[6];
  const float* Wl2 = (const float*)d_in[7];
  const float* bl2 = (const float*)d_in[8];
  const float* Wr2 = (const float*)d_in[9];
  const float* Wl3 = (const float*)d_in[10];
  const float* bl3 = (const float*)d_in[11];
  const float* Wr3 = (const float*)d_in[12];
  const float* Wl4 = (const float*)d_in[13];
  const float* bl4 = (const float*)d_in[14];
  const float* Wr4 = (const float*)d_in[15];
  const float* Prel1  = (const float*)d_in[16];
  const float* prb1   = (const float*)d_in[17];
  const float* Proot1 = (const float*)d_in[18];
  const float* Prel2  = (const float*)d_in[19];
  const float* prb2   = (const float*)d_in[20];
  const float* Proot2 = (const float*)d_in[21];
  const float* W5 = (const float*)d_in[22];
  const float* b5 = (const float*)d_in[23];
  const float* W6 = (const float*)d_in[24];
  const float* b6 = (const float*)d_in[25];

  float* ws    = (float*)d_ws;
  float* agg   = ws;              // Nt*128
  float* hA    = agg + NtF;       // Nt*128
  float* hB    = hA + NtF;        // Nt*128
  float* cnt   = hB + NtF;        // Nt
  float* nm    = cnt + Nt;        // Nt
  float* score = nm + Nt;         // Nt
  float* em    = score + Nt;      // E
  float* gbuf  = em + E;          // G*128
  // packed f16 weights (8 matrices x 16384 halfs), 64B aligned
  _Float16* wpk = (_Float16*)((((uintptr_t)(gbuf + (size_t)G * HID)) + 63u) & ~(uintptr_t)63u);
  _Float16* Wl1p = wpk + 0 * 16384;
  _Float16* Wr1p = wpk + 1 * 16384;
  _Float16* Wl2p = wpk + 2 * 16384;
  _Float16* Wr2p = wpk + 3 * 16384;
  _Float16* Wl3p = wpk + 4 * 16384;
  _Float16* Wr3p = wpk + 5 * 16384;
  _Float16* Wl4p = wpk + 6 * 16384;
  _Float16* Wr4p = wpk + 7 * 16384;

  const long aggThreads = (long)E * 32;

  // pack all weights once
  k_pack_w<<<64, 256, 0, stream>>>(Wl1, Wl1p);
  k_pack_w<<<64, 256, 0, stream>>>(Wr1, Wr1p);
  k_pack_w<<<64, 256, 0, stream>>>(Wl2, Wl2p);
  k_pack_w<<<64, 256, 0, stream>>>(Wr2, Wr2p);
  k_pack_w<<<64, 256, 0, stream>>>(Wl3, Wl3p);
  k_pack_w<<<64, 256, 0, stream>>>(Wr3, Wr3p);
  k_pack_w<<<64, 256, 0, stream>>>(Wl4, Wl4p);
  k_pack_w<<<64, 256, 0, stream>>>(Wr4, Wr4p);

  k_fill<<<cdiv(E, 256), 256, 0, stream>>>(em, 1.f, E);
  k_fill<<<cdiv(Nt, 256), 256, 0, stream>>>(nm, 1.f, Nt);

  // ---- SAGE layer 1: x -> hA
  k_fill<<<cdiv(NtF, 256), 256, 0, stream>>>(agg, 0.f, NtF);
  k_fill<<<cdiv(Nt, 256), 256, 0, stream>>>(cnt, 0.f, Nt);
  k_aggregate<<<cdiv(aggThreads, 256), 256, 0, stream>>>(x, src, dst, em, agg, cnt, E, 1);
  k_mean<<<cdiv(NtF, 256), 256, 0, stream>>>(agg, cnt, NtF);
  k_sage_gemm<<<Nt / 16, 256, 0, stream>>>(agg, x, Wl1p, Wr1p, bl1, nm, hA);

  // ---- SAGE layer 2: hA -> hB
  k_fill<<<cdiv(NtF, 256), 256, 0, stream>>>(agg, 0.f, NtF);
  k_fill<<<cdiv(Nt, 256), 256, 0, stream>>>(cnt, 0.f, Nt);
  k_aggregate<<<cdiv(aggThreads, 256), 256, 0, stream>>>(hA, src, dst, em, agg, cnt, E, 1);
  k_mean<<<cdiv(NtF, 256), 256, 0, stream>>>(agg, cnt, NtF);
  k_sage_gemm<<<Nt / 16, 256, 0, stream>>>(agg, hA, Wl2p, Wr2p, bl2, nm, hB);

  // ---- SAGPool 1 (k = 1024), acts on hB
  k_fill<<<cdiv(NtF, 256), 256, 0, stream>>>(agg, 0.f, NtF);
  k_aggregate<<<cdiv(aggThreads, 256), 256, 0, stream>>>(hB, src, dst, em, agg, cnt, E, 0);
  k_score<<<cdiv((long)Nt * 32, 256), 256, 0, stream>>>(agg, hB, Prel1, Proot1, prb1, score, Nt);
  k_topk<<<G, 1024, 0, stream>>>(score, nm, 1024);
  k_pool_h<<<cdiv(NtF, 256), 256, 0, stream>>>(hB, score, nm, NtF);
  k_pool_em<<<cdiv(E, 256), 256, 0, stream>>>(em, src, dst, nm, E);

  // ---- SAGE layer 3: hB -> hA (masked)
  k_fill<<<cdiv(NtF, 256), 256, 0, stream>>>(agg, 0.f, NtF);
  k_fill<<<cdiv(Nt, 256), 256, 0, stream>>>(cnt, 0.f, Nt);
  k_aggregate<<<cdiv(aggThreads, 256), 256, 0, stream>>>(hB, src, dst, em, agg, cnt, E, 1);
  k_mean<<<cdiv(NtF, 256), 256, 0, stream>>>(agg, cnt, NtF);
  k_sage_gemm<<<Nt / 16, 256, 0, stream>>>(agg, hB, Wl3p, Wr3p, bl3, nm, hA);

  // ---- SAGE layer 4: hA -> hB (masked)
  k_fill<<<cdiv(NtF, 256), 256, 0, stream>>>(agg, 0.f, NtF);
  k_fill<<<cdiv(Nt, 256), 256, 0, stream>>>(cnt, 0.f, Nt);
  k_aggregate<<<cdiv(aggThreads, 256), 256, 0, stream>>>(hA, src, dst, em, agg, cnt, E, 1);
  k_mean<<<cdiv(NtF, 256), 256, 0, stream>>>(agg, cnt, NtF);
  k_sage_gemm<<<Nt / 16, 256, 0, stream>>>(agg, hA, Wl4p, Wr4p, bl4, nm, hB);

  // ---- SAGPool 2 (k = 512), acts on hB
  k_fill<<<cdiv(NtF, 256), 256, 0, stream>>>(agg, 0.f, NtF);
  k_aggregate<<<cdiv(aggThreads, 256), 256, 0, stream>>>(hB, src, dst, em, agg, cnt, E, 0);
  k_score<<<cdiv((long)Nt * 32, 256), 256, 0, stream>>>(agg, hB, Prel2, Proot2, prb2, score, Nt);
  k_topk<<<G, 1024, 0, stream>>>(score, nm, 512);
  k_pool_h<<<cdiv(NtF, 256), 256, 0, stream>>>(hB, score, nm, NtF);
  // em no longer needed after this point

  // ---- readout + head
  k_readout<<<G, HID, 0, stream>>>(hB, nm, gbuf);
  k_head<<<G, 64, 0, stream>>>(gbuf, W5, b5, W6, b6, (float*)d_out);
}
